// Model_12962211299361
// MI455X (gfx1250) — matrix operand order repaired and checked by reading, not compiled, not run on hardware
//
#include <hip/hip_runtime.h>

#define NN  8192
#define DIM 128

typedef __attribute__((ext_vector_type(16))) __bf16 v16bf;
typedef __attribute__((ext_vector_type(8)))  __bf16 v8bf;
typedef __attribute__((ext_vector_type(8)))  float  v8f;
typedef __attribute__((ext_vector_type(4)))  float  v4f;
typedef __attribute__((ext_vector_type(4)))  int    v4i;

#define CHUNK_ELEMS 4096   // 8 frags * 512 bf16 = 8KB per 32-col chunk

// ---------------------------------------------------------------------------
// Async staging of one 8KB B chunk into LDS (256 threads, 32B each),
// tracked by ASYNCcnt (gfx1250 global_load_async_to_lds_b128).
// ---------------------------------------------------------------------------
#if defined(__has_builtin)
#if __has_builtin(__builtin_amdgcn_global_load_async_to_lds_b128)
#define HAVE_ASYNC_LDS 1
#endif
#endif

#ifdef HAVE_ASYNC_LDS
typedef __attribute__((address_space(1))) v4i gv4i;   // global int4
typedef __attribute__((address_space(3))) v4i sv4i;   // LDS int4
#endif

__device__ __forceinline__ void stage_chunk(__bf16* dstLds,
                                            const __bf16* srcGlobal, int tid) {
#ifdef HAVE_ASYNC_LDS
    __builtin_amdgcn_global_load_async_to_lds_b128(
        (gv4i*)(srcGlobal + tid * 8), (sv4i*)(dstLds + tid * 8), 0, 0);
    __builtin_amdgcn_global_load_async_to_lds_b128(
        (gv4i*)(srcGlobal + 2048 + tid * 8), (sv4i*)(dstLds + 2048 + tid * 8), 0, 0);
#else
    v8bf t0 = *(const v8bf*)(srcGlobal + tid * 8);
    v8bf t1 = *(const v8bf*)(srcGlobal + 2048 + tid * 8);
    *(v8bf*)(dstLds + tid * 8) = t0;
    *(v8bf*)(dstLds + 2048 + tid * 8) = t1;
#endif
}

__device__ __forceinline__ void wait_stage() {
#ifdef HAVE_ASYNC_LDS
#if __has_builtin(__builtin_amdgcn_s_wait_asynccnt)
    __builtin_amdgcn_s_wait_asynccnt(0);
#else
    asm volatile("s_wait_asynccnt 0x0" ::: "memory");
#endif
#endif
}

__device__ __forceinline__ void load_a_chunk(v4f aq[4], const float* arow,
                                             int jc, int hi) {
    const float* p0 = arow + (jc << 5) + 8 * hi;
    aq[0] = __builtin_nontemporal_load((const v4f*)p0);
    aq[1] = __builtin_nontemporal_load((const v4f*)(p0 + 4));
    aq[2] = __builtin_nontemporal_load((const v4f*)(p0 + 16));
    aq[3] = __builtin_nontemporal_load((const v4f*)(p0 + 20));
}

// ---------------------------------------------------------------------------
// Kernel 1: V = X @ W^T + b (f32 math), stored bf16 in WMMA-B swizzled layout:
// frag*512 + (e>>3)*256 + lane*8 + (e&7), lane = n + 16*hi, k = 8*hi + e (e<8), 16 + 8*hi + (e-8) (e>=8).
// ---------------------------------------------------------------------------
__global__ void xw_swizzle_kernel(const float* __restrict__ X,
                                  const float* __restrict__ W,
                                  const float* __restrict__ bias,
                                  __bf16* __restrict__ Vfrag) {
    __shared__ float xs[DIM];
    const int i = blockIdx.x;
    const int o = threadIdx.x;
    xs[o] = X[(size_t)i * DIM + o];
    __syncthreads();

    const float* wr = W + (size_t)o * DIM;
    float acc = bias[o];
#pragma unroll 8
    for (int d = 0; d < DIM; ++d) acc += xs[d] * wr[d];

    const int jc = i >> 5, kk = i & 31;
    const int t = o >> 4, n = o & 15;
    const int hi = (kk >> 3) & 1, e = (kk & 7) + ((kk >> 4) << 3);
    const int lane = n + 16 * hi;
    const size_t off = ((size_t)(jc * 8 + t)) * 512 + (size_t)(e >> 3) * 256
                     + (size_t)lane * 8 + (e & 7);
    Vfrag[off] = (__bf16)acc;
}

// ---------------------------------------------------------------------------
// Kernel 2: fused masked-softmax @ V, split-K partials.
// Workgroup = 8 waves = 128 rows sharing one K segment; the 8KB B chunk is
// double-buffered in LDS via async copies. One wave per 16-row tile.
// ---------------------------------------------------------------------------
__global__ void __launch_bounds__(256)
gcn_fused_kernel(const float* __restrict__ A,
                 const __bf16* __restrict__ Vfrag,
                 float* __restrict__ pout,   // [S][NN][DIM]
                 float* __restrict__ prs,    // [S][NN]
                 int chunks_per_seg) {
    __shared__ __bf16 lds[2 * CHUNK_ELEMS];   // 2 x 8KB double buffer

    const int tid  = threadIdx.x;
    const int wave = tid >> 5;
    const int lane = tid & 31;
    const int hi   = lane >> 4;
    const int m    = lane & 15;
    const int tile = blockIdx.x * 8 + wave;   // 16-row tile, 0..511
    const int seg  = blockIdx.y;
    const int row  = tile * 16 + m;
    const float* __restrict__ arow = A + (size_t)row * NN;
    const int jc0 = seg * chunks_per_seg;

    v8f   acc[8] = {};
    float ls[4]  = {0.f, 0.f, 0.f, 0.f};

    // Consume one staged chunk: masked exp -> bf16 A-fragment, then 8 WMMAs
    // with the B-fragment LDS loads software-pipelined one fragment ahead.
    auto compute = [&](const v4f aq[4], const __bf16* fb) {
        v16bf afrag;
#pragma unroll
        for (int q = 0; q < 4; ++q) {
#pragma unroll
            for (int k = 0; k < 4; ++k) {
                const float a  = aq[q][k];
                const float pv = (a > 0.8f) ? __expf(a) : 0.0f;   // logits <= 1
                ls[k] += pv;
                afrag[q * 4 + k] = (__bf16)pv;
            }
        }
        v8bf blo = *(const v8bf*)(fb + lane * 8);
        v8bf bhi = *(const v8bf*)(fb + 256 + lane * 8);
#pragma unroll
        for (int t = 0; t < 8; ++t) {
            v8bf nlo = blo, nhi = bhi;
            if (t < 7) {
                nlo = *(const v8bf*)(fb + (t + 1) * 512 + lane * 8);
                nhi = *(const v8bf*)(fb + (t + 1) * 512 + 256 + lane * 8);
            }
            v16bf bfrag = __builtin_shufflevector(
                blo, bhi, 0, 1, 2, 3, 4, 5, 6, 7, 8, 9, 10, 11, 12, 13, 14, 15);
            acc[t] = __builtin_amdgcn_wmma_f32_16x16x32_bf16(
                false, afrag, false, bfrag, (short)0, acc[t], false, false);
            blo = nlo; bhi = nhi;
        }
    };

    // ---- prologue: chunk 0 -> buf0 / aqA
    v4f aqA[4], aqB[4];
    stage_chunk(&lds[0], Vfrag + (size_t)jc0 * CHUNK_ELEMS, tid);
    load_a_chunk(aqA, arow, jc0, hi);
    wait_stage();
    __syncthreads();

    // chunks_per_seg is always even (32/64/128/256): unroll by 2 so the
    // double buffers alternate statically (no register copies, no 'cur').
    const int halves = chunks_per_seg >> 1;
    for (int c2 = 0; c2 < halves; ++c2) {
        const int c = 2 * c2;

        // even chunk: consume buf0/aqA, prefetch c+1 -> buf1/aqB
        stage_chunk(&lds[CHUNK_ELEMS], Vfrag + (size_t)(jc0 + c + 1) * CHUNK_ELEMS, tid);
        load_a_chunk(aqB, arow, jc0 + c + 1, hi);
        compute(aqA, &lds[0]);
        wait_stage();
        __syncthreads();

        // odd chunk: consume buf1/aqB, prefetch c+2 -> buf0/aqA
        if (c + 2 < chunks_per_seg) {
            stage_chunk(&lds[0], Vfrag + (size_t)(jc0 + c + 2) * CHUNK_ELEMS, tid);
            load_a_chunk(aqA, arow, jc0 + c + 2, hi);
        }
        compute(aqB, &lds[CHUNK_ELEMS]);
        wait_stage();
        __syncthreads();
    }

    // ---- full row sum: lanes L and L^16 cover disjoint K halves of row m
    const float lanesum = (ls[0] + ls[1]) + (ls[2] + ls[3]);
    const float rs_full = lanesum + __shfl_xor(lanesum, 16, 32);
    if (hi == 0)
        prs[(size_t)seg * NN + row] = rs_full;

    // ---- store unnormalized partial tile
    float* po = pout + ((size_t)seg * NN + (size_t)tile * 16) * DIM;
#pragma unroll
    for (int t = 0; t < 8; ++t) {
#pragma unroll
        for (int v = 0; v < 8; ++v) {
            const int r = v + 8 * hi;
            po[(size_t)r * DIM + t * 16 + m] = acc[t][v];
        }
    }
}

// ---------------------------------------------------------------------------
// Kernel 3: reduce split-K partials, normalize by row sum, leaky-relu.
// ---------------------------------------------------------------------------
__global__ void finalize_kernel(const float* __restrict__ pout,
                                const float* __restrict__ prs,
                                float* __restrict__ out, int S) {
    const int i = blockIdx.x;
    const int o = threadIdx.x;
    float rs = 0.0f;
    for (int s = 0; s < S; ++s) rs += prs[(size_t)s * NN + i];
    float acc = 0.0f;
    for (int s = 0; s < S; ++s)
        acc += pout[((size_t)s * NN + i) * DIM + o];
    const float inv = (rs > 0.0f) ? (1.0f / rs) : 0.0f;
    const float v = acc * inv;
    out[(size_t)i * DIM + o] = (v > 0.0f) ? v : 0.01f * v;
}

// ---------------------------------------------------------------------------
extern "C" void kernel_launch(void* const* d_in, const int* in_sizes, int n_in,
                              void* d_out, int out_size, void* d_ws, size_t ws_size,
                              hipStream_t stream) {
    const float* A  = (const float*)d_in[0];
    const float* X  = (const float*)d_in[1];
    const float* W  = (const float*)d_in[2];
    const float* b  = (const float*)d_in[3];
    float* out = (float*)d_out;

    const size_t vfrag_bytes = (size_t)NN * DIM * sizeof(__bf16);   // 2 MB
    const size_t seg_bytes   = (size_t)NN * DIM * sizeof(float)
                             + (size_t)NN * sizeof(float);

    // Split-K factor for occupancy; deterministic in ws_size only.
    int S = 1;
    if      (ws_size >= vfrag_bytes + 8 * seg_bytes) S = 8;
    else if (ws_size >= vfrag_bytes + 4 * seg_bytes) S = 4;
    else if (ws_size >= vfrag_bytes + 2 * seg_bytes) S = 2;

    __bf16* Vfrag = (__bf16*)d_ws;
    float*  pout;
    float*  prs;
    if (ws_size >= vfrag_bytes + (size_t)S * seg_bytes) {
        pout = (float*)((char*)d_ws + vfrag_bytes);
        prs  = pout + (size_t)S * NN * DIM;
    } else {
        // Minimal-workspace fallback: S==1, accumulate into d_out.
        S = 1;
        pout = out;
        prs  = (float*)((char*)d_ws + vfrag_bytes);
    }

    // 1) V = bf16_swizzled(X @ W^T + b)
    xw_swizzle_kernel<<<NN, DIM, 0, stream>>>(X, W, b, Vfrag);

    // 2) fused masked-softmax @ V (8 row-tiles per workgroup share LDS chunk)
    dim3 grid(NN / DIM, S);
    gcn_fused_kernel<<<grid, 256, 0, stream>>>(A, Vfrag, pout, prs, 256 / S);

    // 3) reduce + normalize + leaky-relu
    finalize_kernel<<<NN, DIM, 0, stream>>>(pout, prs, out, S);
}
